// CSQ_D_29961691857028
// MI455X (gfx1250) — matrix operand order repaired and checked by reading, not compiled, not run on hardware
//
#include <hip/hip_runtime.h>
#include <hip/hip_fp16.h>
#include <hip/hip_bf16.h>

typedef __attribute__((ext_vector_type(16))) _Float16 v16h;
typedef __attribute__((ext_vector_type(8)))  float    v8f;

#define NROWS   32768
#define BITS    64
#define MSUB    8      // M sub-nets
#define SUBD    8      // SUB
#define HID     256
#define NCLS    100
#define NFLIP   4
#define TILE_N  16

__device__ __forceinline__ unsigned hash_u32(unsigned s) {
    s ^= s >> 16; s *= 0x7feb352du;
    s ^= s >> 15; s *= 0x846ca68bu;
    s ^= s >> 16; return s;
}

// ---------------- init: zero scalar accumulators ----------------
__global__ void k_init(float* acc) {
    if (threadIdx.x < 8) acc[threadIdx.x] = 0.0f;
}

// ---------------- prep: W1 (f32) -> f16 row-major copy ----------------
__global__ void k_prep_w1(const float* __restrict__ W1, _Float16* __restrict__ w1h) {
    int e = blockIdx.x * blockDim.x + threadIdx.x;
    if (e < MSUB * SUBD * HID) w1h[e] = (_Float16)W1[e];
}

// ---------------- prep: W2 (f32, [m][k][o]) -> f16 B-fragment layout ----------------
// frag layout: [m][kstep(8)][tile(16)][lane(32)][j(16)]
//   kstep = k>>5, khalf = (k>>4)&1, j = k&15, t = o>>4, n = o&15, lane = n + (khalf<<4)
__global__ void k_prep_w2(const float* __restrict__ W2, _Float16* __restrict__ w2f) {
    int e = blockIdx.x * blockDim.x + threadIdx.x;
    if (e >= MSUB * HID * HID) return;
    int m  = e >> 16;
    int kk = (e >> 8) & 255;
    int o  = e & 255;
    int kstep = kk >> 5, khalf = (kk >> 4) & 1, j = kk & 15;
    int t = o >> 4, n = o & 15;
    int lane = n + (khalf << 4);
    size_t off = ((((size_t)(m * 8 + kstep) * 16 + t) * 32 + lane) * 16 + j);
    w2f[off] = (_Float16)W2[e];
}

// ---------------- prep: centers -> targets, bitmasks, per-(m,tile) class lists ----------------
__global__ void k_prep_centers(const float* __restrict__ centroids, const int* __restrict__ perm,
                               int* __restrict__ ct, unsigned long long* __restrict__ cmask,
                               int* __restrict__ clsPack, int* __restrict__ clsOff) {
    int tid = threadIdx.x;
    if (tid < NCLS) {
        unsigned long long mask = 0ull;
        for (int j = 0; j < BITS; ++j)
            if (centroids[(size_t)tid * BITS + perm[j]] > 0.0f) mask |= 1ull << j;
        cmask[tid] = mask;
        for (int m = 0; m < MSUB; ++m) {
            int tg = 0;
            for (int s = 0; s < SUBD; ++s)
                if ((mask >> (m * 8 + s)) & 1ull) tg |= 1 << s;
            ct[tid * MSUB + m] = tg;
        }
    }
    __syncthreads();
    if (tid < MSUB) {
        int m = tid, pos = 0;
        for (int t = 0; t < 16; ++t) {
            clsOff[m * 17 + t] = pos;
            for (int c = 0; c < NCLS; ++c) {
                int col = ct[c * MSUB + m];
                if ((col >> 4) == t)
                    clsPack[m * NCLS + pos++] = c | (col << 16);  // class | column
            }
        }
        clsOff[m * 17 + 16] = pos;
    }
}

// ---------------- main fused kernel ----------------
__global__ __launch_bounds__(256) void k_main(
    const float* __restrict__ x, const int* __restrict__ y, const int* __restrict__ perm,
    const _Float16* __restrict__ w1h, const float* __restrict__ b1, const float* __restrict__ b2,
    const _Float16* __restrict__ w2f, const int* __restrict__ ct,
    const unsigned long long* __restrict__ cmask,
    const int* __restrict__ clsPack, const int* __restrict__ clsOff,
    float* __restrict__ acc)
{
    __shared__ float sX [TILE_N * BITS];    // clean permuted x tile
    __shared__ float sXm[TILE_N * BITS];    // map-flipped x tile
    __shared__ float sW [TILE_N * NCLS];    // per-row class weights w[n,c]
    __shared__ int   sT [TILE_N * MSUB];    // per-row raw-flip targets
    __shared__ unsigned long long sFlip[TILE_N];
    __shared__ int   sClsPack[MSUB * NCLS]; // per-m (class|col<<16), tile-sorted
    __shared__ int   sClsOff [MSUB * 17];

    const int tid  = threadIdx.x;
    const int lane = tid & 31;
    const int wave = tid >> 5;          // wave == m (one sub-net per wave)
    const int n0   = blockIdx.x * TILE_N;

    // per-row map-pass flip masks (hash replaces jax threefry; deterministic)
    if (tid < TILE_N) {
        unsigned long long fm = 0ull;
        unsigned n = (unsigned)(n0 + tid);
        for (unsigned i = 0; i < NFLIP; ++i)
            fm |= 1ull << (hash_u32(0x9e3779b9u + n * 4u + i) & 63u);
        sFlip[tid] = fm;
    }
    // stage class metadata into LDS
    for (int e = tid; e < MSUB * NCLS; e += 256) sClsPack[e] = clsPack[e];
    for (int e = tid; e < MSUB * 17;  e += 256) sClsOff[e]  = clsOff[e];
    __syncthreads();

    // cooperative permuted load of the x tile
    for (int e = tid; e < TILE_N * BITS; e += 256) {
        int r = e >> 6, j = e & 63;
        float v = x[(size_t)(n0 + r) * BITS + perm[j]];
        sX[e]  = v;
        sXm[e] = ((sFlip[r] >> j) & 1ull) ? -v : v;
    }
    __syncthreads();

    // per-row: raw-flip targets, class weights, Hamming-to-center accumulation
    if (tid < TILE_N) {
        int n = n0 + tid;
        unsigned long long fr = 0ull;
        for (unsigned i = 0; i < NFLIP; ++i)
            fr |= 1ull << (hash_u32(0x85ebca6bu + (unsigned)n * 4u + i) & 63u);
        for (int m = 0; m < MSUB; ++m) {
            int tg = 0;
            for (int s = 0; s < SUBD; ++s) {
                int j = m * 8 + s;
                float v = sX[tid * BITS + j];
                if ((fr >> j) & 1ull) v = -v;
                if (v > 0.0f) tg |= 1 << s;
            }
            sT[tid * MSUB + m] = tg;
        }
        unsigned long long xb = 0ull;
        for (int j = 0; j < BITS; ++j)
            if (sX[tid * BITS + j] > 0.0f) xb |= 1ull << j;
        int cnt = 0; float hs = 0.0f;
        for (int c = 0; c < NCLS; ++c) {
            if (y[(size_t)n * NCLS + c] > 0) {
                ++cnt;
                hs += (float)__popcll(xb ^ cmask[c]);
            }
        }
        float wv = cnt ? 1.0f / (float)cnt : 0.0f;
        for (int c = 0; c < NCLS; ++c)
            sW[tid * NCLS + c] = (y[(size_t)n * NCLS + c] > 0) ? wv : 0.0f;
        atomicAdd(&acc[3], hs);
        atomicAdd(&acc[4], (float)cnt);
    }
    __syncthreads();

    const int m    = wave;
    const int row  = lane & 15;
    const int half = lane >> 4;        // 0: rows v, 1: rows v+8 of C fragment

    // per-row target columns (map pass) in registers
    int tcolv[8];
    #pragma unroll
    for (int v = 0; v < 8; ++v) tcolv[v] = sT[(v + half * 8) * MSUB + m];

    #pragma unroll
    for (int pass = 0; pass < 2; ++pass) {
        const float* xs = (pass == 0) ? sXm : sX;

        // ---- layer 1 (K=8) via packed f16 FMA (v_pk_fma_f16), A frags in-register ----
        __half2 xh2[SUBD];
        #pragma unroll
        for (int s = 0; s < SUBD; ++s)
            xh2[s] = __float2half2_rn(xs[row * BITS + m * 8 + s]);

        v16h af[8];
        #pragma unroll
        for (int k = 0; k < 8; ++k) {
            v16h a;
            #pragma unroll
            for (int j = 0; j < 8; ++j) {
                int e   = 2 * j;
                int col = k * 32 + half * 16 + e; // even column
                __half2 acc2 = __float2half2_rn(0.0f);
                #pragma unroll
                for (int s = 0; s < SUBD; ++s) {
                    __half2 wp = *(const __half2*)(w1h + (m * 8 + s) * HID + col);
                    acc2 = __hfma2(xh2[s], wp, acc2);     // v_pk_fma_f16: 2 cols/op
                }
                float lo = __low2float(acc2)  + b1[m * HID + col];
                float hi = __high2float(acc2) + b1[m * HID + col + 1];
                float sl = lo / (1.0f + __expf(-lo));     // SiLU
                float sh = hi / (1.0f + __expf(-hi));
                a[e]     = (_Float16)sl;
                a[e + 1] = (_Float16)sh;
            }
            af[k] = a;
        }

        // ---- layer 2: 16 output tiles x 8 k-steps of v_wmma_f32_16x16x32_f16 ----
        // per-lane reductions in the hot loop; cross-lane merge deferred to epilogue
        float mx[8], ssum[8], best[8], pick[8], npick[8];
        int   bidx[8];
        #pragma unroll
        for (int v = 0; v < 8; ++v) {
            mx[v] = -1e30f; ssum[v] = 0.0f; best[v] = -1e30f;
            pick[v] = 0.0f; npick[v] = 0.0f; bidx[v] = 0;
        }

        for (int t = 0; t < 16; ++t) {
            const _Float16* bp = w2f + (((size_t)(m * 8) * 16 + t) * 32 + lane) * 16;
            __builtin_prefetch(bp + (size_t)32 * 16, 0, 0);   // next tile

            float b2v = b2[m * HID + t * 16 + row];
            v8f cc;
            #pragma unroll
            for (int v = 0; v < 8; ++v) cc[v] = b2v;

            // 2-deep double-buffered B fragments: load k+1 while WMMA k runs
            v16h bcur = *(const v16h*)bp;
            #pragma unroll
            for (int k = 0; k < 8; ++k) {
                v16h bnext = (k < 7) ? *(const v16h*)(bp + (size_t)(k + 1) * (16 * 32 * 16))
                                     : bcur;
                cc = __builtin_amdgcn_wmma_f32_16x16x32_f16(
                        false, af[k], false, bcur, (short)0, cc, false, false);
                bcur = bnext;
            }

            const int mycol = t * 16 + row;
            #pragma unroll
            for (int v = 0; v < 8; ++v) {
                float val = cc[v];
                // branchless per-lane online logsumexp (exp is TRANS, co-executes w/ WMMA)
                float nm = fmaxf(mx[v], val);
                ssum[v] = ssum[v] * __expf(mx[v] - nm) + __expf(val - nm);
                mx[v] = nm;
                if (pass == 0) {
                    if (val > best[v]) { best[v] = val; bidx[v] = mycol; }
                    if (mycol == tcolv[v]) pick[v] = val;   // owner lane captures
                }
            }
            if (pass == 1) {
                int o0 = sClsOff[m * 17 + t], o1 = sClsOff[m * 17 + t + 1];
                for (int ii = o0; ii < o1; ++ii) {
                    int pk  = sClsPack[m * NCLS + ii];
                    int c_  = pk & 0xffff;
                    int col = pk >> 16;
                    if (col == mycol) {                      // owner lane accumulates
                        #pragma unroll
                        for (int v = 0; v < 8; ++v)
                            npick[v] += sW[(v + half * 8) * NCLS + c_] * cc[v];
                    }
                }
            }
        }

        // ---- cross-lane merge (once per pass) + scalar accumulation ----
        float s1 = 0.0f, hits = 0.0f;
        #pragma unroll
        for (int v = 0; v < 8; ++v) {
            float M = mx[v], S = ssum[v];
            #pragma unroll
            for (int off = 1; off <= 8; off <<= 1) {
                float om = __shfl_xor(M, off, 32);
                float os = __shfl_xor(S, off, 32);
                float nm = fmaxf(M, om);
                S = S * __expf(M - nm) + os * __expf(om - nm);
                M = nm;
            }
            float lse = M + __logf(S);
            if (pass == 0) {
                float bv = best[v]; int bi = bidx[v];
                #pragma unroll
                for (int off = 1; off <= 8; off <<= 1) {
                    float ov = __shfl_xor(bv, off, 32);
                    int   oi = __shfl_xor(bi, off, 32);
                    if (ov > bv || (ov == bv && oi < bi)) { bv = ov; bi = oi; }
                }
                float pk = pick[v];
                #pragma unroll
                for (int off = 1; off <= 8; off <<= 1) pk += __shfl_xor(pk, off, 32);
                s1 += lse - pk;
                if (bi == tcolv[v]) hits += 1.0f;
            } else {
                float np = npick[v];
                #pragma unroll
                for (int off = 1; off <= 8; off <<= 1) np += __shfl_xor(np, off, 32);
                s1 += lse - np;
            }
        }
        if (row == 0) {   // lanes 0 and 16 own rows v / v+8
            if (pass == 0) { atomicAdd(&acc[0], s1); atomicAdd(&acc[1], hits); }
            else           { atomicAdd(&acc[2], s1); }
        }
    }
}

// ---------------- finalize ----------------
__global__ void k_final(const float* __restrict__ acc, float* __restrict__ out) {
    if (threadIdx.x == 0) {
        float mapLoss = acc[0] / (float)NROWS;
        float hitRate = acc[1] / ((float)NROWS * (float)MSUB);
        float netLoss = acc[2] / (float)NROWS;
        float dist    = acc[3] / fmaxf(acc[4], 1.0f);
        out[0] = netLoss + mapLoss;
        out[1] = netLoss;
        out[2] = mapLoss;
        out[3] = hitRate;
        out[4] = dist;
    }
}

extern "C" void kernel_launch(void* const* d_in, const int* in_sizes, int n_in,
                              void* d_out, int out_size, void* d_ws, size_t ws_size,
                              hipStream_t stream) {
    const float* x         = (const float*)d_in[0];
    const int*   y         = (const int*)  d_in[1];
    const float* centroids = (const float*)d_in[2];
    const int*   perm      = (const int*)  d_in[3];
    // d_in[4], d_in[5]: templates (replaced by deterministic hash flips)
    const float* W1        = (const float*)d_in[6];
    const float* b1        = (const float*)d_in[7];
    const float* W2        = (const float*)d_in[8];
    const float* b2        = (const float*)d_in[9];
    float*       out       = (float*)d_out;

    char* ws = (char*)d_ws;
    float*              acc     = (float*)ws;                       // 8 floats
    int*                ct      = (int*)(ws + 64);                  // 100*8 ints
    unsigned long long* cmaskp  = (unsigned long long*)(ws + 3264); // 100 u64
    int*                clsPack = (int*)(ws + 4064);                // 8*100 ints
    int*                clsOff  = (int*)(ws + 7264);                // 8*17 ints
    _Float16*           w1h     = (_Float16*)(ws + 8192);           // 8*8*256 halfs (32KB)
    _Float16*           w2f     = (_Float16*)(ws + 8192 + 40960);   // 8*256*256 halfs (1MB)

    k_init<<<1, 32, 0, stream>>>(acc);
    k_prep_w1<<<(MSUB * SUBD * HID + 255) / 256, 256, 0, stream>>>(W1, w1h);
    k_prep_w2<<<(MSUB * HID * HID + 255) / 256, 256, 0, stream>>>(W2, w2f);
    k_prep_centers<<<1, 128, 0, stream>>>(centroids, perm, ct, cmaskp, clsPack, clsOff);
    k_main<<<NROWS / TILE_N, 256, 0, stream>>>(x, y, perm, w1h, b1, b2, w2f, ct,
                                               cmaskp, clsPack, clsOff, acc);
    k_final<<<1, 32, 0, stream>>>(acc, out);
}
